// MultiHeadSelfAttention_8426725834917
// MI455X (gfx1250) — matrix operand order repaired and checked by reading, not compiled, not run on hardware
//
#include <hip/hip_runtime.h>

typedef _Float16 v16h __attribute__((ext_vector_type(16)));
typedef _Float16 v8h  __attribute__((ext_vector_type(8)));
typedef float    v8f  __attribute__((ext_vector_type(8)));

__device__ __forceinline__ v16h mk16(v8h lo, v8h hi) {
  v16h r;
#pragma unroll
  for (int i = 0; i < 8; ++i) { r[i] = lo[i]; r[i + 8] = hi[i]; }
  return r;
}

// 16-byte global -> LDS copy. On gfx1250 use the async-tensor data path
// (ASYNCcnt-tracked, overlaps with WMMA); else synchronous fallback.
__device__ __forceinline__ void copy16_g2l(const _Float16* g, _Float16* l) {
#if defined(__gfx1250__)
  unsigned loff = (unsigned)(size_t)l;              // LDS aperture: addr[31:0]
  unsigned long long ga = (unsigned long long)g;
  asm volatile("global_load_async_to_lds_b128 %0, %1, off"
               :: "v"(loff), "v"(ga) : "memory");
#else
  *(v8h*)l = *(const v8h*)g;
#endif
}
__device__ __forceinline__ void async_wait0() {
#if defined(__gfx1250__)
  asm volatile("s_wait_asynccnt 0x0" ::: "memory");
#endif
}

// ---------------------------------------------------------------- fp32 -> f16
__global__ void cvt_f32_f16(const float* __restrict__ in,
                            _Float16* __restrict__ out, int n) {
  int i = blockIdx.x * blockDim.x + threadIdx.x;
  if (i < n) out[i] = (_Float16)in[i];
}

// ---------------------------------------------------------------- WMMA GEMM
// D[M x N] = A[M x K] * Bw[N x K]^T + bias.
// Block (256 thr = 8 waves) owns a 128(M) x 64(N) tile; per 32-wide K-step the
// block stages A(128x32) and B(64x32) f16 tiles into LDS with async copies
// (double buffered), then each wave runs 4 WMMAs on its 16x64 sub-tile.
// LDS row stride 40 halves (80 B): 16-B aligned b128 loads, conflict-free.
#define KS 40
__global__ __launch_bounds__(256) void gemm_wmma_lds(
    const _Float16* __restrict__ A, const _Float16* __restrict__ Bw,
    const float* __restrict__ bias, _Float16* __restrict__ outH,
    float* __restrict__ outF, int M, int N, int K) {
  __shared__ __align__(16) _Float16 As[2][128 * KS];
  __shared__ __align__(16) _Float16 Bs[2][64 * KS];

  int tid = threadIdx.x;
  int wave = tid >> 5, lane = tid & 31;
  int li = lane & 15, hi = lane >> 4;

  int ntiles = N >> 6;
  int mt = blockIdx.x / ntiles;
  int nt = blockIdx.x - mt * ntiles;

  const _Float16* Ag = A + (size_t)(mt * 128) * K;
  const _Float16* Bg = Bw + (size_t)(nt * 64) * K;

  // copy assignments: 16-B chunks; A tile = 512 chunks (2/thread), B = 256 (1/thread)
  int ar0 = tid >> 2;                 // rows 0..63
  int ar1 = ar0 + 64;                 // rows 64..127
  int ak  = (tid & 3) * 8;            // k offset in halves (0,8,16,24)
  int br  = tid >> 2;                 // B rows 0..63

  v8f acc[4];
  v8f zero = {};
#pragma unroll
  for (int j = 0; j < 4; ++j) acc[j] = zero;

  // prologue: stage K-step 0 into buffer 0
  copy16_g2l(Ag + (size_t)ar0 * K + ak, &As[0][ar0 * KS + ak]);
  copy16_g2l(Ag + (size_t)ar1 * K + ak, &As[0][ar1 * KS + ak]);
  copy16_g2l(Bg + (size_t)br  * K + ak, &Bs[0][br  * KS + ak]);
  async_wait0();
  __syncthreads();

  for (int kt = 0; kt < K; kt += 32) {
    int cur = (kt >> 5) & 1, nxt = cur ^ 1;
    if (kt + 32 < K) {                 // overlap next stage with this compute
      int kn = kt + 32;
      copy16_g2l(Ag + (size_t)ar0 * K + kn + ak, &As[nxt][ar0 * KS + ak]);
      copy16_g2l(Ag + (size_t)ar1 * K + kn + ak, &As[nxt][ar1 * KS + ak]);
      copy16_g2l(Bg + (size_t)br  * K + kn + ak, &Bs[nxt][br  * KS + ak]);
    }

    const _Float16* Asl = &As[cur][(wave * 16 + li) * KS + hi * 8];
    v16h a = mk16(*(const v8h*)(Asl), *(const v8h*)(Asl + 16));
#pragma unroll
    for (int j = 0; j < 4; ++j) {
      const _Float16* Bsl = &Bs[cur][(j * 16 + li) * KS + hi * 8];
      v16h b = mk16(*(const v8h*)(Bsl), *(const v8h*)(Bsl + 16));
      acc[j] = __builtin_amdgcn_wmma_f32_16x16x32_f16(
          false, a, false, b, (short)0, acc[j], false, false);
    }

    async_wait0();      // next buffer staged
    __syncthreads();    // + this buffer's ds reads done before overwrite
  }

#pragma unroll
  for (int j = 0; j < 4; ++j) {
    int col = nt * 64 + j * 16 + li;
    float bv = bias ? bias[col] : 0.0f;
#pragma unroll
    for (int v = 0; v < 8; ++v) {
      int row = mt * 128 + wave * 16 + v + hi * 8;
      float val = acc[j][v] + bv;
      size_t idx = (size_t)row * N + col;
      if (outH) outH[idx] = (_Float16)val;
      else      outF[idx] = val;
    }
  }
}

// ------------------------------------------------- per-token head attention
// One wave per token t = b*L + l. qkv row: q[h*64+d], k at +1024, v at +2048.
// S = Q K^T * 1/8 (2 WMMAs, K=64) -> row softmax -> P@V (4 WMMAs, K padded
// 16->32 with zeros) -> f16 store into the swapaxes+reshape-scrambled layout:
// dest row = b*4096 + h*256 + l/16, col = (l%16)*64 + d.
__global__ void attn_heads_wmma(const _Float16* __restrict__ qkv,
                                _Float16* __restrict__ outS) {
  __shared__ float Ps[16][17];
  int t = blockIdx.x;
  int lane = threadIdx.x & 31;
  int li = lane & 15, hi = lane >> 4;
  const _Float16* Q  = qkv + (size_t)t * 3072;
  const _Float16* Kh = Q + 1024;
  const _Float16* Vh = Q + 2048;

  v8f s = {};
#pragma unroll
  for (int kt = 0; kt < 64; kt += 32) {
    int ka = kt + hi * 8;
    v16h a = mk16(*(const v8h*)(Q + li * 64 + ka),
                  *(const v8h*)(Q + li * 64 + ka + 16));
    int kb = kt + hi * 8;
    v16h b = mk16(*(const v8h*)(Kh + li * 64 + kb),
                  *(const v8h*)(Kh + li * 64 + kb + 16));
    s = __builtin_amdgcn_wmma_f32_16x16x32_f16(
        false, a, false, b, (short)0, s, false, false);
  }

  const float scale = 0.125f;  // d^-0.5, d = 64
#pragma unroll
  for (int v = 0; v < 8; ++v) {
    float x = s[v] * scale;
    float m = x;
#pragma unroll
    for (int msk = 1; msk < 16; msk <<= 1)
      m = fmaxf(m, __shfl_xor(m, msk, 32));
    float e = __expf(x - m);
    float sum = e;
#pragma unroll
    for (int msk = 1; msk < 16; msk <<= 1)
      sum += __shfl_xor(sum, msk, 32);
    Ps[v + hi * 8][li] = e / sum;
  }
  __syncthreads();

  v16h a2;
#pragma unroll
  for (int e = 0; e < 16; ++e) {
    a2[e] = hi ? (_Float16)0.0f : (_Float16)Ps[li][e];
  }

  v8f zero = {};
  int bidx = t >> 12;        // L = 4096
  int l = t & 4095;
  size_t rbase = (size_t)bidx * 4096 + (size_t)(l >> 4);
  size_t cbase = (size_t)(l & 15) * 64;
#pragma unroll
  for (int j = 0; j < 4; ++j) {
    v16h b;
#pragma unroll
    for (int e = 0; e < 16; ++e) {
      _Float16 val = Vh[(size_t)e * 64 + j * 16 + li];  // V[g=e][d]
      b[e] = hi ? (_Float16)0.0f : val;                 // K >= 16 -> zero
    }
    v8f o = __builtin_amdgcn_wmma_f32_16x16x32_f16(
        false, a2, false, b, (short)0, zero, false, false);
    int d = j * 16 + li;
#pragma unroll
    for (int v = 0; v < 8; ++v) {
      int h = v + hi * 8;
      size_t row = rbase + (size_t)h * 256;
      outS[row * 1024 + cbase + d] = (_Float16)o[v];
    }
  }
}

// ----------------------------------------------------------------- launcher
extern "C" void kernel_launch(void* const* d_in, const int* in_sizes, int n_in,
                              void* d_out, int out_size, void* d_ws, size_t ws_size,
                              hipStream_t stream) {
  const float* x    = (const float*)d_in[0];
  const float* Wqkv = (const float*)d_in[1];
  const float* bqkv = (const float*)d_in[2];
  const float* Wout = (const float*)d_in[3];
  const float* bout = (const float*)d_in[4];
  float* out = (float*)d_out;

  const int M  = 4 * 4096;   // B*L tokens
  const int C  = 1024;
  const int N1 = 3 * C;

  _Float16* xh    = (_Float16*)d_ws;
  _Float16* wqkvh = xh    + (size_t)M * C;
  _Float16* wouth = wqkvh + (size_t)N1 * C;
  _Float16* qkvh  = wouth + (size_t)C * C;
  _Float16* attnh = qkvh  + (size_t)M * N1;

  int n1 = M * C;
  cvt_f32_f16<<<(n1 + 255) / 256, 256, 0, stream>>>(x, xh, n1);
  int n2 = N1 * C;
  cvt_f32_f16<<<(n2 + 255) / 256, 256, 0, stream>>>(Wqkv, wqkvh, n2);
  int n3 = C * C;
  cvt_f32_f16<<<(n3 + 255) / 256, 256, 0, stream>>>(Wout, wouth, n3);

  // qkv = x @ Wqkv^T + b  (f16 out)
  int blocks1 = (M / 128) * (N1 / 64);                  // 6144
  gemm_wmma_lds<<<blocks1, 256, 0, stream>>>(xh, wqkvh, bqkv,
                                             qkvh, nullptr, M, N1, C);

  // per-token 16-head attention + scramble store
  attn_heads_wmma<<<M, 32, 0, stream>>>(qkvh, attnh);

  // out = attn_scrambled @ Wout^T + b_out (fp32 out)
  int blocks2 = (M / 128) * (C / 64);                   // 2048
  gemm_wmma_lds<<<blocks2, 256, 0, stream>>>(attnh, wouth, bout,
                                             nullptr, out, M, C, C);
}
